// MultiHeadAttn_33449205301807
// MI455X (gfx1250) — matrix operand order repaired and checked by reading, not compiled, not run on hardware
//
#include <hip/hip_runtime.h>

// ---------------------------------------------------------------------------
// MultiHeadAttention forward for MI455X (gfx1250, wave32, WMMA + async-LDS).
// Pipeline:
//   0) cvt_f32_f16: one-shot f32->f16 of q/k/v and Wq/Wk/Wv/Wo
//   1) gemm16<0/1/2>: Y = X @ W^T + b, double-buffered async-LDS staging
//        mode 0: Q -> [B,H,S,DK] scaled by log2(e)/sqrt(DK) (base-2 softmax)
//        mode 1: K -> [B,H,S,DK]
//        mode 2: V -> [B,H,DK,S] (transposed for contiguous P@V B-frags)
//   2) flash_attn: 64-key tiles, online base-2 softmax, row-sums via WMMA
//   3) gemm16<3>: out = Xc @ Wo^T + bo (f32 result)
// ---------------------------------------------------------------------------

typedef __attribute__((ext_vector_type(16))) _Float16 v16h;
typedef __attribute__((ext_vector_type(8)))  _Float16 h8;
typedef __attribute__((ext_vector_type(4)))  _Float16 h4;
typedef __attribute__((ext_vector_type(8)))  float    v8f;

#define DEV static __device__ __forceinline__

DEV v16h cat16(h8 lo, h8 hi) {
  return __builtin_shufflevector(lo, hi, 0, 1, 2, 3, 4, 5, 6, 7,
                                 8, 9, 10, 11, 12, 13, 14, 15);
}

DEV v8f wmma16(v16h a, v16h b, v8f c) {
  // D = A(16x32 f16) * B(32x16 f16) + C(16x16 f32)
  return __builtin_amdgcn_wmma_f32_16x16x32_f16(false, a, false, b,
                                                (short)0, c, false, false);
}

DEV v8f vzero8() {
  v8f z;
#pragma unroll
  for (int i = 0; i < 8; ++i) z[i] = 0.0f;
  return z;
}

// Async 16-byte global -> LDS copy (per lane), tracked by ASYNCcnt.
// Low 32 bits of a generic pointer to __shared__ are the LDS byte address.
DEV void async_ld16(void* lds, const void* g) {
  asm volatile("global_load_async_to_lds_b128 %0, %1, off"
               :: "v"((unsigned)(unsigned long long)lds), "v"(g)
               : "memory");
}
DEV void wait_async_le4() { asm volatile("s_wait_asynccnt 0x4" ::: "memory"); }

static constexpr int Bc = 4, Sc = 2048, Dc = 1024, Hc = 16, DKc = 64;
// softmax done base-2: fold log2(e)/sqrt(DK) into Q at projection time
static constexpr float kQScale = 0.125f * 1.44269504088896340736f;

// ---------------------------------------------------------------------------
// One-shot f32 -> f16 conversion (float4 in, h4 out).
// ---------------------------------------------------------------------------
__global__ __launch_bounds__(256) void cvt_f32_f16(const float* __restrict__ in,
                                                   _Float16* __restrict__ out,
                                                   int n4) {
  const int i = blockIdx.x * 256 + threadIdx.x;
  if (i < n4) {
    float4 v = ((const float4*)in)[i];
    h4 o = {(_Float16)v.x, (_Float16)v.y, (_Float16)v.z, (_Float16)v.w};
    ((h4*)out)[i] = o;
  }
}

// ---------------------------------------------------------------------------
// GEMM: Y[m,n] = sum_k X[m,k] * W[n,k] + bias[n]   (X, W already f16)
// Block tile 128x128, BK=32; 8 waves, each 32x64 (2x4 WMMA frags).
// Double-buffered async-LDS staging; s_wait_asynccnt 4 retires the older tile.
// ---------------------------------------------------------------------------
template <int MODE>  // 0=Q (scaled), 1=K, 2=V (transposed), 3=final f32 out
__global__ __launch_bounds__(256) void gemm16(const _Float16* __restrict__ X,
                                              const _Float16* __restrict__ W,
                                              const float* __restrict__ bias,
                                              void* __restrict__ Yv) {
  __shared__ alignas(16) _Float16 As[2][128][40];  // +8 pad: conflict-free
  __shared__ alignas(16) _Float16 Bs[2][128][40];

  const int tid = threadIdx.x;
  const int wid = tid >> 5, lane = tid & 31;
  const int ln = lane & 15, lh = lane >> 4;
  const int wm = (wid & 3) << 5;   // wave M offset: 0,32,64,96
  const int wn = (wid >> 2) << 6;  // wave N offset: 0,64
  const int m0 = blockIdx.y << 7;
  const int n0 = blockIdx.x << 7;

  // Per-thread staging chunks: A/B tiles are 128 rows x 32 f16 = 512 x 16B.
  const int r0 = tid >> 2, c0 = (tid & 3) << 3;
  const int r1 = (tid + 256) >> 2, c1 = ((tid + 256) & 3) << 3;

  auto stage = [&](int buf, int kt) {
    async_ld16(&As[buf][r0][c0], X + (size_t)(m0 + r0) * Dc + kt + c0);
    async_ld16(&Bs[buf][r0][c0], W + (size_t)(n0 + r0) * Dc + kt + c0);
    async_ld16(&As[buf][r1][c1], X + (size_t)(m0 + r1) * Dc + kt + c1);
    async_ld16(&Bs[buf][r1][c1], W + (size_t)(n0 + r1) * Dc + kt + c1);
  };

  v8f acc[2][4];
#pragma unroll
  for (int i = 0; i < 2; ++i)
#pragma unroll
    for (int j = 0; j < 4; ++j) acc[i][j] = vzero8();

  stage(0, 0);
  for (int t = 0; t < Dc / 32; ++t) {
    const int cur = t & 1;
    stage(cur ^ 1, ((t + 1) & (Dc / 32 - 1)) << 5);  // prefetch (wraps on last)
    wait_async_le4();   // older tile's 4 copies have landed
    __syncthreads();

    v16h af[2], bf[4];
#pragma unroll
    for (int i = 0; i < 2; ++i) {
      const _Float16* r = &As[cur][wm + i * 16 + ln][0];
      af[i] = cat16(*(const h8*)(r + 16 * lh), *(const h8*)(r + 16 * lh + 8));
    }
#pragma unroll
    for (int j = 0; j < 4; ++j) {
      const _Float16* r = &Bs[cur][wn + j * 16 + ln][0];
      bf[j] = cat16(*(const h8*)(r + 16 * lh), *(const h8*)(r + 16 * lh + 8));
    }
#pragma unroll
    for (int i = 0; i < 2; ++i)
#pragma unroll
      for (int j = 0; j < 4; ++j) acc[i][j] = wmma16(af[i], bf[j], acc[i][j]);

    __syncthreads();  // all waves done reading buf `cur` before it is refilled
  }

  // Epilogue: bias, scale, head split / transpose / f32 store.
#pragma unroll
  for (int i = 0; i < 2; ++i)
#pragma unroll
    for (int j = 0; j < 4; ++j) {
      const int n = n0 + wn + j * 16 + ln;
      const float bn = bias[n];
      const int hh = n >> 6, dk = n & 63;
#pragma unroll
      for (int r = 0; r < 8; ++r) {
        const int m = m0 + wm + i * 16 + r + 8 * lh;
        const float val = acc[i][j][r] + bn;
        if constexpr (MODE == 3) {
          ((float*)Yv)[(size_t)m * Dc + n] = val;
        } else {
          _Float16* Y = (_Float16*)Yv;
          const int b = m >> 11, s = m & (Sc - 1);
          if constexpr (MODE == 0)
            Y[(((size_t)b * Hc + hh) * Sc + s) * DKc + dk] =
                (_Float16)(val * kQScale);
          else if constexpr (MODE == 1)
            Y[(((size_t)b * Hc + hh) * Sc + s) * DKc + dk] = (_Float16)val;
          else
            Y[(((size_t)b * Hc + hh) * DKc + dk) * Sc + s] = (_Float16)val;
        }
      }
    }
}

// ---------------------------------------------------------------------------
// Flash attention, 64-key tiles. Grid: (S/128, B*H). 8 waves x 16 query rows.
// Q A-frags in registers for the whole sweep; K/V tiles double-buffered via
// async copies (4 per wave per tile). Base-2 online softmax; row max via
// 4-step xor-shuffle, row sum via WMMA against an all-ones B matrix.
// ---------------------------------------------------------------------------
__global__ __launch_bounds__(256) void flash_attn(const _Float16* __restrict__ Qh,
                                                  const _Float16* __restrict__ Kh,
                                                  const _Float16* __restrict__ Vt,
                                                  _Float16* __restrict__ Xc) {
  const int bh = blockIdx.y;          // b*H + h
  const int b = bh >> 4, h = bh & 15;
  const int wid = threadIdx.x >> 5, lane = threadIdx.x & 31;
  const int ln = lane & 15, lh = lane >> 4;

  const int qrow = (blockIdx.x << 7) + wid * 16 + ln;
  const _Float16* Qrow = Qh + ((size_t)bh * Sc + qrow) * DKc;

  // Q A-frags: k 0..31 and 32..63 (pre-scaled by log2(e)/sqrt(DK))
  v16h aq0 = cat16(*(const h8*)(Qrow + 16 * lh), *(const h8*)(Qrow + 16 * lh + 8));
  v16h aq1 = cat16(*(const h8*)(Qrow + 32 + 16 * lh), *(const h8*)(Qrow + 40 + 16 * lh));

  v16h ones;
#pragma unroll
  for (int i = 0; i < 16; ++i) ones[i] = (_Float16)1.0f;

  __shared__ alignas(16) _Float16 Ks[2][64][72];   // [key][dk], pad 72
  __shared__ alignas(16) _Float16 Vs[2][64][72];   // [dk][key], pad 72
  __shared__ alignas(16) _Float16 Ps[8][16][72];   // per-wave P transpose

  const _Float16* Kbase = Kh + (size_t)bh * Sc * DKc;
  const _Float16* Vbase = Vt + (size_t)bh * DKc * Sc;

  // K/V tiles are 64 rows x 64 f16 = 512 x 16B chunks; 2 K + 2 V per thread.
  const int sr0 = threadIdx.x >> 3, sc0 = (threadIdx.x & 7) << 3;
  const int sr1 = (threadIdx.x + 256) >> 3, sc1 = ((threadIdx.x + 256) & 7) << 3;
  auto stage = [&](int buf, int kb) {
    async_ld16(&Ks[buf][sr0][sc0], Kbase + (size_t)(kb + sr0) * DKc + sc0);
    async_ld16(&Vs[buf][sr0][sc0], Vbase + (size_t)sr0 * Sc + kb + sc0);
    async_ld16(&Ks[buf][sr1][sc1], Kbase + (size_t)(kb + sr1) * DKc + sc1);
    async_ld16(&Vs[buf][sr1][sc1], Vbase + (size_t)sr1 * Sc + kb + sc1);
  };

  float mrow[8];
  v8f ls = vzero8();   // row sums (replicated across lanes, C-layout rows)
  v8f o[4];
#pragma unroll
  for (int r = 0; r < 8; ++r) mrow[r] = -3.0e38f;
#pragma unroll
  for (int t = 0; t < 4; ++t) o[t] = vzero8();

  stage(0, 0);
  for (int t = 0; t < Sc / 64; ++t) {
    const int cur = t & 1;
    stage(cur ^ 1, ((t + 1) & (Sc / 64 - 1)) << 6);  // prefetch next 64 keys
    wait_async_le4();  // current tile's 4 copies have landed
    __syncthreads();

    // scores for 64 keys: four 16-key C-frags, each reduced over DK=64
    v8f c[4];
#pragma unroll
    for (int g = 0; g < 4; ++g) {
      const _Float16* kr = &Ks[cur][g * 16 + ln][0];
      v16h bk0 = cat16(*(const h8*)(kr + 16 * lh), *(const h8*)(kr + 16 * lh + 8));
      v16h bk1 = cat16(*(const h8*)(kr + 32 + 16 * lh), *(const h8*)(kr + 40 + 16 * lh));
      c[g] = wmma16(aq0, bk0, vzero8());
      c[g] = wmma16(aq1, bk1, c[g]);
    }

    // online softmax (base 2); rows live in 16-lane halves -> xor 1,2,4,8
#pragma unroll
    for (int r = 0; r < 8; ++r) {
      float bm = fmaxf(fmaxf(c[0][r], c[1][r]), fmaxf(c[2][r], c[3][r]));
      bm = fmaxf(bm, __shfl_xor(bm, 1));
      bm = fmaxf(bm, __shfl_xor(bm, 2));
      bm = fmaxf(bm, __shfl_xor(bm, 4));
      bm = fmaxf(bm, __shfl_xor(bm, 8));
      const float mnew = fmaxf(mrow[r], bm);
      const float alpha = __builtin_amdgcn_exp2f(mrow[r] - mnew);
      mrow[r] = mnew;
      ls[r] *= alpha;
#pragma unroll
      for (int tt = 0; tt < 4; ++tt) o[tt][r] *= alpha;
      // P = 2^(s - mnew), straight into the per-wave transpose buffer
#pragma unroll
      for (int g = 0; g < 4; ++g)
        Ps[wid][r + 8 * lh][g * 16 + ln] =
            (_Float16)__builtin_amdgcn_exp2f(c[g][r] - mnew);
    }
    asm volatile("s_wait_dscnt 0" ::: "memory");  // same-wave DS RAW fence

    // P A-frags for kj 0..31 and 32..63
    const _Float16* pr = &Ps[wid][ln][0];
    v16h ap0 = cat16(*(const h8*)(pr + 16 * lh), *(const h8*)(pr + 16 * lh + 8));
    v16h ap1 = cat16(*(const h8*)(pr + 32 + 16 * lh), *(const h8*)(pr + 40 + 16 * lh));

    // row sums: ls += P @ ones  (replaces the shuffle-sum reduction)
    ls = wmma16(ap0, ones, ls);
    ls = wmma16(ap1, ones, ls);

    // O += P(16x64) @ V(64x64), 4 DK-tiles x 2 kj-halves
#pragma unroll
    for (int tt = 0; tt < 4; ++tt) {
      const _Float16* vr = &Vs[cur][tt * 16 + ln][0];
      v16h bv0 = cat16(*(const h8*)(vr + 16 * lh), *(const h8*)(vr + 16 * lh + 8));
      v16h bv1 = cat16(*(const h8*)(vr + 32 + 16 * lh), *(const h8*)(vr + 40 + 16 * lh));
      o[tt] = wmma16(ap0, bv0, o[tt]);
      o[tt] = wmma16(ap1, bv1, o[tt]);
    }

    __syncthreads();  // all waves done with buf `cur` before it is refilled
  }

  // normalize by row sums (ls replicated across lanes) and scatter to [B,S,D]
#pragma unroll
  for (int t = 0; t < 4; ++t)
#pragma unroll
    for (int r = 0; r < 8; ++r) {
      const int qi = (blockIdx.x << 7) + wid * 16 + r + 8 * lh;
      const int d = h * DKc + t * 16 + ln;
      Xc[((size_t)b * Sc + qi) * Dc + d] = (_Float16)(o[t][r] / ls[r]);
    }
}

// ---------------------------------------------------------------------------
extern "C" void kernel_launch(void* const* d_in, const int* in_sizes, int n_in,
                              void* d_out, int out_size, void* d_ws, size_t ws_size,
                              hipStream_t stream) {
  (void)in_sizes; (void)n_in; (void)out_size; (void)ws_size;

  const float* q  = (const float*)d_in[0];
  const float* k  = (const float*)d_in[1];
  const float* v  = (const float*)d_in[2];
  const float* Wq = (const float*)d_in[3];
  const float* bq = (const float*)d_in[4];
  const float* Wk = (const float*)d_in[5];
  const float* bk = (const float*)d_in[6];
  const float* Wv = (const float*)d_in[7];
  const float* bv = (const float*)d_in[8];
  const float* Wo = (const float*)d_in[9];
  const float* bo = (const float*)d_in[10];

  const size_t xe = (size_t)Bc * Sc * Dc;  // 8.39M
  const size_t we = (size_t)Dc * Dc;       // 1.05M
  _Float16* Xq16 = (_Float16*)d_ws;        // later reused as Xc (stream order)
  _Float16* Xk16 = Xq16 + xe;
  _Float16* Xv16 = Xk16 + xe;
  _Float16* Qh   = Xv16 + xe;
  _Float16* Kh   = Qh + xe;
  _Float16* Vt   = Kh + xe;
  _Float16* Wq16 = Vt + xe;
  _Float16* Wk16 = Wq16 + we;
  _Float16* Wv16 = Wk16 + we;
  _Float16* Wo16 = Wv16 + we;
  _Float16* Xc   = Xq16;  // alias: gemm16<0> consumed Xq16 before attention

  const int xb = (int)(xe / 4 / 256);  // 8192 blocks
  const int wb = (int)(we / 4 / 256);  // 1024 blocks
  cvt_f32_f16<<<xb, 256, 0, stream>>>(q, Xq16, (int)(xe / 4));
  cvt_f32_f16<<<xb, 256, 0, stream>>>(k, Xk16, (int)(xe / 4));
  cvt_f32_f16<<<xb, 256, 0, stream>>>(v, Xv16, (int)(xe / 4));
  cvt_f32_f16<<<wb, 256, 0, stream>>>(Wq, Wq16, (int)(we / 4));
  cvt_f32_f16<<<wb, 256, 0, stream>>>(Wk, Wk16, (int)(we / 4));
  cvt_f32_f16<<<wb, 256, 0, stream>>>(Wv, Wv16, (int)(we / 4));
  cvt_f32_f16<<<wb, 256, 0, stream>>>(Wo, Wo16, (int)(we / 4));

  dim3 gg(Dc / 128, (Bc * Sc) / 128);  // 8 x 64
  gemm16<0><<<gg, 256, 0, stream>>>(Xq16, Wq16, bq, Qh);
  gemm16<1><<<gg, 256, 0, stream>>>(Xk16, Wk16, bk, Kh);
  gemm16<2><<<gg, 256, 0, stream>>>(Xv16, Wv16, bv, Vt);

  flash_attn<<<dim3(Sc / 128, Bc * Hc), 256, 0, stream>>>(Qh, Kh, Vt, Xc);

  gemm16<3><<<gg, 256, 0, stream>>>(Xc, Wo16, bo, d_out);
}
